// SuperLoss1_18983755448979
// MI455X (gfx1250) — matrix operand order repaired and checked by reading, not compiled, not run on hardware
//
#include <hip/hip_runtime.h>
#include <hip/hip_bf16.h>

#define BDIM 384
#define KDIM 512
#define TAU_F 1.3862943611198906f
#define LAM_F 0.25f
#define MARGIN_F 0.1f

typedef _Float16 v16h __attribute__((ext_vector_type(16)));
typedef _Float16 v8h  __attribute__((ext_vector_type(8)));
typedef float    v8f  __attribute__((ext_vector_type(8)));

// ---------------- fp32 -> f16 conversion ----------------
__global__ void cvt_to_f16(const float* __restrict__ x, _Float16* __restrict__ xh, int n) {
    int i = blockIdx.x * blockDim.x + threadIdx.x;
    if (i < n) xh[i] = (_Float16)x[i];
}

// ---------------- Gram matrix G = X * X^T via WMMA ----------------
// One wave32 per 16x16 output tile. K=512 -> 16 x (16x16x32) WMMA steps.
__global__ void gram_wmma(const _Float16* __restrict__ Xh, float* __restrict__ G) {
    const int tilesPerRow = BDIM / 16;                       // 24
    int wave = blockIdx.x * (blockDim.x >> 5) + (threadIdx.x >> 5);
    if (wave >= tilesPerRow * tilesPerRow) return;
    int lane = threadIdx.x & 31;
    int tm = wave / tilesPerRow;
    int tn = wave % tilesPerRow;
    int hi  = lane >> 4;     // half-wave select
    int l16 = lane & 15;

    // A-matrix: lane (l16) = row M; 16 contiguous K starting at hi*16
    const _Float16* arow = Xh + (size_t)(tm * 16 + l16) * KDIM;
    // B-matrix: lane (l16) = col N; 16 contiguous K starting at hi*16 (row of X = column of X^T)
    const _Float16* brow = Xh + (size_t)(tn * 16 + l16) * KDIM;

    v8f c = {};
#pragma unroll 4
    for (int k0 = 0; k0 < KDIM; k0 += 32) {
        v16h a = *(const v16h*)(arow + k0 + hi * 16);
        v16h b = *(const v16h*)(brow + k0 + hi * 16);
        // (neg_a, A, neg_b, B, c_mod, C, reuse_a, reuse_b)
        c = __builtin_amdgcn_wmma_f32_16x16x32_f16(false, a, false, b, (short)0, c,
                                                   false, false);
    }
    // C/D layout: VGPR r -> M = hi*8 + r, N = l16
    float* gout = G + (size_t)(tm * 16 + hi * 8) * BDIM + tn * 16 + l16;
#pragma unroll
    for (int r = 0; r < 8; ++r) gout[(size_t)r * BDIM] = c[r];
}

// ---------------- SuperLoss sigma terms ----------------
__device__ __forceinline__ float lambertw_f(float y) {
    const float Ec = 2.718281828459045f;
    float p = sqrtf(fmaxf(2.f * (Ec * y + 1.f), 0.f));
    float w = (y < 0.25f) ? (-1.f + p * (1.f - p * (1.f / 3.f)))
                          : logf(fmaxf(y, 0.25f));
#pragma unroll
    for (int it = 0; it < 12; ++it) {
        float ew = expf(w);
        float f  = w * ew - y;
        w = w - f / (ew * (w + 1.f) - (w + 2.f) * f / (2.f * w + 2.f) + 1e-12f);
    }
    return w;
}

// (l - tau)*sigma + lam*log(sigma)^2 with sigma = exp(-W(y)), log(sigma) = -W
__device__ __forceinline__ float loss_term(float l) {
    const float neg2overE = -0.7357588823428847f;
    float y = 0.5f * fmaxf(neg2overE, (l - TAU_F) / LAM_F);
    float w = lambertw_f(y);
    float s = expf(-w);
    return (l - TAU_F) * s + LAM_F * w * w;
}

// From G build D, then Ap[i,j] = term(D), Bn[i,j] = term(relu(margin - D))
__global__ void make_terms(const float* __restrict__ G,
                           float* __restrict__ Ap, float* __restrict__ Bn) {
    int idx = blockIdx.x * blockDim.x + threadIdx.x;
    if (idx >= BDIM * BDIM) return;
    int i = idx / BDIM, j = idx % BDIM;
    float sqi = G[(size_t)i * BDIM + i];
    float sqj = G[(size_t)j * BDIM + j];
    float d2  = fmaxf(sqi - 2.f * G[idx] + sqj, 0.f);
    float D   = sqrtf(d2 + 1e-16f);
    Ap[idx] = loss_term(D);
    float ln = fmaxf(MARGIN_F - D, 0.f);
    Bn[idx] = loss_term(ln);
}

// ---------------- O(B^3) masked reduction ----------------
// Block = anchor i. Valid j: j!=i && lab[j]==lab[i]. Valid k: lab[k]!=lab[i]
// (this also excludes k==i and k==j). Deterministic block-tree reduction.
__global__ void triplet_sum(const float* __restrict__ Ap, const float* __restrict__ Bn,
                            const int* __restrict__ lab,
                            float* __restrict__ partS, float* __restrict__ partC) {
    __shared__ float sBn[BDIM];
    __shared__ int   sLab[BDIM];
    __shared__ float redS[128];
    __shared__ float redC[128];
    int i = blockIdx.x;
    for (int t = threadIdx.x; t < BDIM; t += blockDim.x) {
        sBn[t]  = Bn[(size_t)i * BDIM + t];
        sLab[t] = lab[t];
    }
    __syncthreads();
    int li = sLab[i];
    float sum = 0.f, cnt = 0.f;
    for (int j = threadIdx.x; j < BDIM; j += blockDim.x) {
        if (j == i || sLab[j] != li) continue;
        float aij = Ap[(size_t)i * BDIM + j];
        for (int k = 0; k < BDIM; ++k) {
            if (sLab[k] == li) continue;
            float v = fmaxf(aij + sBn[k], 0.f);
            sum += v;
            if (v > 1e-16f) cnt += 1.f;
        }
    }
    redS[threadIdx.x] = sum;
    redC[threadIdx.x] = cnt;
    __syncthreads();
    for (int off = blockDim.x >> 1; off > 0; off >>= 1) {
        if ((int)threadIdx.x < off) {
            redS[threadIdx.x] += redS[threadIdx.x + off];
            redC[threadIdx.x] += redC[threadIdx.x + off];
        }
        __syncthreads();
    }
    if (threadIdx.x == 0) { partS[i] = redS[0]; partC[i] = redC[0]; }
}

__global__ void finalize_k(const float* __restrict__ partS,
                           const float* __restrict__ partC,
                           float* __restrict__ out) {
    if (threadIdx.x == 0 && blockIdx.x == 0) {
        float s = 0.f, c = 0.f;
        for (int i = 0; i < BDIM; ++i) { s += partS[i]; c += partC[i]; }
        out[0] = s / (c + 1e-16f);
    }
}

extern "C" void kernel_launch(void* const* d_in, const int* in_sizes, int n_in,
                              void* d_out, int out_size, void* d_ws, size_t ws_size,
                              hipStream_t stream) {
    (void)in_sizes; (void)n_in; (void)out_size; (void)ws_size;
    const float* logits  = (const float*)d_in[0];
    const int*   targets = (const int*)d_in[1];
    float*       out     = (float*)d_out;

    char* ws = (char*)d_ws;
    const size_t XH_BYTES = (size_t)BDIM * KDIM * sizeof(_Float16);  // 393216
    const size_t M_BYTES  = (size_t)BDIM * BDIM * sizeof(float);     // 589824
    _Float16* Xh   = (_Float16*)ws;
    float*    G    = (float*)(ws + XH_BYTES);
    float*    Ap   = (float*)(ws + XH_BYTES + M_BYTES);
    float*    Bn   = (float*)(ws + XH_BYTES + 2 * M_BYTES);
    float*    partS = (float*)(ws + XH_BYTES + 3 * M_BYTES);
    float*    partC = partS + BDIM;

    int n = BDIM * KDIM;
    cvt_to_f16<<<(n + 255) / 256, 256, 0, stream>>>(logits, Xh, n);

    const int tiles = (BDIM / 16) * (BDIM / 16);                 // 576 waves
    const int wavesPerBlock = 8;                                 // 256 threads
    gram_wmma<<<(tiles + wavesPerBlock - 1) / wavesPerBlock, 256, 0, stream>>>(Xh, G);

    make_terms<<<(BDIM * BDIM + 255) / 256, 256, 0, stream>>>(G, Ap, Bn);

    triplet_sum<<<BDIM, 128, 0, stream>>>(Ap, Bn, targets, partS, partC);

    finalize_k<<<1, 1, 0, stream>>>(partS, partC, out);
}
